// MultiHeadAttention_9689446220512
// MI455X (gfx1250) — compile-verified, matrix operand order checked by reading, not run on hardware
//
#include <hip/hip_runtime.h>

// ---------------------------------------------------------------------------
// MHA fused pipeline for gfx1250 (CDNA5, wave32, WMMA 16x16x32 bf16)
// Explicit ping-pong double buffering: zero register-copy rotation, loads
// overlap WMMAs via partial s_wait_loadcnt.
// ---------------------------------------------------------------------------

typedef __bf16 bf16;
typedef __attribute__((ext_vector_type(16))) __bf16 v16bf;
typedef __attribute__((ext_vector_type(8)))  __bf16 v8bf;
typedef __attribute__((ext_vector_type(8)))  float  v8f;

union ABf { v16bf v; v8bf h[2]; };

#define D_MODEL 1024
#define QKV_N   3072
#define SEQ     2048
#define NHEAD   16
#define DHEAD   64
#define ROWS    8192   // 4 * 2048

__device__ __forceinline__ v8f wmma_bf16(const ABf& a, const ABf& b, v8f c) {
    return __builtin_amdgcn_wmma_f32_16x16x32_bf16(
        /*neg_a=*/false, a.v, /*neg_b=*/false, b.v,
        /*c_mod=*/(short)0, c, /*reuse_a=*/false, /*reuse_b=*/false);
}

__device__ __forceinline__ v8f vzero8() {
    v8f z = {0.f, 0.f, 0.f, 0.f, 0.f, 0.f, 0.f, 0.f};
    return z;
}

// load a 16x32 bf16 WMMA operand (this lane's two 16-byte halves)
__device__ __forceinline__ void load_op(const bf16* p, ABf& a) {
    a.h[0] = *(const v8bf*)(p);
    a.h[1] = *(const v8bf*)(p + 16);
}

// ---------------------------------------------------------------------------
// Shared 16(M) x 64(N) x 1024(K) register-tile GEMM body, ping-pong buffered.
// aptr: this lane's A-operand base; bptrN: this lane's B bases for 4 N-tiles.
// ---------------------------------------------------------------------------
__device__ __forceinline__ void gemm_16x64_pipe(
    const bf16* __restrict__ aptr,
    const bf16* __restrict__ bptr0, const bf16* __restrict__ bptr1,
    const bf16* __restrict__ bptr2, const bf16* __restrict__ bptr3,
    v8f acc[4])
{
    ABf aA, aB;
    ABf bA0, bA1, bA2, bA3;
    ABf bB0, bB1, bB2, bB3;

    load_op(aptr, aA);
    load_op(bptr0, bA0);
    load_op(bptr1, bA1);
    load_op(bptr2, bA2);
    load_op(bptr3, bA3);

    int kk = 0;
#pragma unroll 1
    for (; kk + 64 < D_MODEL; kk += 64) {
        // preload chunk kk+32 into set B, then consume set A (chunk kk)
        load_op(aptr  + kk + 32, aB);
        load_op(bptr0 + kk + 32, bB0);
        load_op(bptr1 + kk + 32, bB1);
        load_op(bptr2 + kk + 32, bB2);
        load_op(bptr3 + kk + 32, bB3);
        acc[0] = wmma_bf16(aA, bA0, acc[0]);
        acc[1] = wmma_bf16(aA, bA1, acc[1]);
        acc[2] = wmma_bf16(aA, bA2, acc[2]);
        acc[3] = wmma_bf16(aA, bA3, acc[3]);
        // preload chunk kk+64 into set A, then consume set B (chunk kk+32)
        load_op(aptr  + kk + 64, aA);
        load_op(bptr0 + kk + 64, bA0);
        load_op(bptr1 + kk + 64, bA1);
        load_op(bptr2 + kk + 64, bA2);
        load_op(bptr3 + kk + 64, bA3);
        acc[0] = wmma_bf16(aB, bB0, acc[0]);
        acc[1] = wmma_bf16(aB, bB1, acc[1]);
        acc[2] = wmma_bf16(aB, bB2, acc[2]);
        acc[3] = wmma_bf16(aB, bB3, acc[3]);
    }
    // tail: chunks kk (in set A) and kk+32 (loaded into set B now)
    load_op(aptr  + kk + 32, aB);
    load_op(bptr0 + kk + 32, bB0);
    load_op(bptr1 + kk + 32, bB1);
    load_op(bptr2 + kk + 32, bB2);
    load_op(bptr3 + kk + 32, bB3);
    acc[0] = wmma_bf16(aA, bA0, acc[0]);
    acc[1] = wmma_bf16(aA, bA1, acc[1]);
    acc[2] = wmma_bf16(aA, bA2, acc[2]);
    acc[3] = wmma_bf16(aA, bA3, acc[3]);
    acc[0] = wmma_bf16(aB, bB0, acc[0]);
    acc[1] = wmma_bf16(aB, bB1, acc[1]);
    acc[2] = wmma_bf16(aB, bB2, acc[2]);
    acc[3] = wmma_bf16(aB, bB3, acc[3]);
}

// ------------------------- conversion kernels ------------------------------

__global__ __launch_bounds__(256) void cvt_f32_bf16(const float* __restrict__ src,
                                                    bf16* __restrict__ dst, int n) {
    int i = blockIdx.x * blockDim.x + threadIdx.x;
    if (i < n) dst[i] = (bf16)src[i];
}

// w is [K, N] row-major f32; wt is [N, K] row-major bf16 (row j = output col j)
__global__ __launch_bounds__(256) void cvt_transpose(const float* __restrict__ w,
                                                     bf16* __restrict__ wt,
                                                     int K, int N) {
    int t = blockIdx.x * blockDim.x + threadIdx.x;
    if (t >= K * N) return;
    int j = t / K;     // output column
    int i = t % K;     // k index (contiguous in wt -> coalesced writes)
    wt[t] = (bf16)w[(size_t)i * N + j];
}

// ------------------------------ QKV GEMM -----------------------------------
// C[8192,3072] = xb @ Wqkv + bqkv; scatter into Q[bh][s][d], K[bh][s][d],
// Vt[bh][d][s] (bf16). Each wave: 16(M) x 64(N) tile.

__global__ __launch_bounds__(256) void qkv_gemm_kernel(
    const bf16* __restrict__ xb, const bf16* __restrict__ wT,
    const float* __restrict__ bias,
    bf16* __restrict__ qb, bf16* __restrict__ kb, bf16* __restrict__ vtb)
{
    const int lane = threadIdx.x & 31;
    const int wave = threadIdx.x >> 5;
    const int half = lane >> 4;
    const int l16  = lane & 15;
    const int m0   = blockIdx.x * 16;
    const int n0   = blockIdx.y * 512 + wave * 64;

    const bf16* aptr  = xb + (size_t)(m0 + l16) * D_MODEL + half * 8;
    const bf16* bptr0 = wT + (size_t)(n0 + 0 * 16 + l16) * D_MODEL + half * 8;
    const bf16* bptr1 = wT + (size_t)(n0 + 1 * 16 + l16) * D_MODEL + half * 8;
    const bf16* bptr2 = wT + (size_t)(n0 + 2 * 16 + l16) * D_MODEL + half * 8;
    const bf16* bptr3 = wT + (size_t)(n0 + 3 * 16 + l16) * D_MODEL + half * 8;

    v8f acc[4];
#pragma unroll
    for (int j = 0; j < 4; ++j) acc[j] = vzero8();

    gemm_16x64_pipe(aptr, bptr0, bptr1, bptr2, bptr3, acc);

#pragma unroll
    for (int j = 0; j < 4; ++j) {
        const int col0  = n0 + j * 16 + l16;
        const int which = col0 >> 10;          // 0=Q 1=K 2=V (uniform per 64-tile)
        const int c1    = col0 & 1023;
        const int h     = c1 >> 6;
        const int d     = c1 & 63;
        const float bcol = bias[col0];
#pragma unroll
        for (int r = 0; r < 8; ++r) {
            const int row = m0 + r + half * 8;
            const int b   = row >> 11;
            const int s   = row & 2047;
            const bf16 bv = (bf16)(acc[j][r] + bcol);
            if (which == 0)
                qb[(((size_t)(b * NHEAD + h) * SEQ + s) * DHEAD) + d] = bv;
            else if (which == 1)
                kb[(((size_t)(b * NHEAD + h) * SEQ + s) * DHEAD) + d] = bv;
            else
                vtb[(((size_t)(b * NHEAD + h) * DHEAD + d) * SEQ) + s] = bv;
        }
    }
}

// ---------------------------- attention kernel -----------------------------
// Flash-attention. Wave handles 16 q-rows; KV tiles of 32; online softmax.
// K operands in distinct registers (partial loadcnt waits); V loads issued
// before softmax so their latency hides behind the softmax VALU work.

__global__ __launch_bounds__(256) void attn_kernel(
    const bf16* __restrict__ qb, const bf16* __restrict__ kb,
    const bf16* __restrict__ vtb, bf16* __restrict__ ob)
{
    __shared__ bf16 plds[8][16 * 32];   // per-wave P tile (C-layout -> A-layout)

    const int lane = threadIdx.x & 31;
    const int wave = threadIdx.x >> 5;
    const int half = lane >> 4;
    const int l16  = lane & 15;
    const int bh   = blockIdx.x;
    const int bi   = bh >> 4;
    const int hi   = bh & 15;
    const int q0   = (blockIdx.y * 8 + wave) * 16;

    const bf16* qh = qb  + (size_t)bh * SEQ * DHEAD;
    const bf16* kh = kb  + (size_t)bh * SEQ * DHEAD;
    const bf16* vh = vtb + (size_t)bh * DHEAD * SEQ;

    // Q A-operands: two 16x32 d-chunks, resident for the whole kv loop
    ABf aq0, aq1;
    {
        const bf16* qp = qh + (size_t)(q0 + l16) * DHEAD + half * 8;
        load_op(qp, aq0);
        load_op(qp + 32, aq1);
    }

    v8f acco[4];
#pragma unroll
    for (int j = 0; j < 4; ++j) acco[j] = vzero8();
    float mrun[8], lrun[8];
#pragma unroll
    for (int r = 0; r < 8; ++r) { mrun[r] = -1e30f; lrun[r] = 0.f; }

    const float scale = 0.125f;   // 1/sqrt(64)

    const bf16* kbase = kh + (size_t)l16 * DHEAD + half * 8;
    const bf16* vbase = vh + (size_t)l16 * SEQ + half * 8;

    for (int kv = 0; kv < SEQ; kv += 32) {
        // ---- K operands: 4 distinct register sets -> partial waits ----
        const bf16* kp  = kbase + (size_t)kv * DHEAD;
        const bf16* kp1 = kp + 16 * DHEAD;
        ABf bk00, bk01, bk10, bk11;
        load_op(kp,       bk00);
        load_op(kp  + 32, bk01);
        load_op(kp1,      bk10);
        load_op(kp1 + 32, bk11);

        v8f s0 = vzero8(), s1 = vzero8();
        s0 = wmma_bf16(aq0, bk00, s0);
        s0 = wmma_bf16(aq1, bk01, s0);
        s1 = wmma_bf16(aq0, bk10, s1);
        s1 = wmma_bf16(aq1, bk11, s1);

        // ---- V operand loads: issue now, consumed after softmax ----
        ABf bv[4];
#pragma unroll
        for (int j = 0; j < 4; ++j)
            load_op(vbase + (size_t)(j * 16) * SEQ + kv, bv[j]);

        if (kv + 32 < SEQ) {
            __builtin_prefetch(kp + 32 * DHEAD, 0, 0);
            __builtin_prefetch(vbase + kv + 32, 0, 0);
        }

        // ---- online softmax (rows spread across the 16 lanes of each half) ----
        float p0[8], p1[8];
#pragma unroll
        for (int r = 0; r < 8; ++r) {
            float v0 = s0[r] * scale;
            float v1 = s1[r] * scale;
            float mx = fmaxf(v0, v1);
            mx = fmaxf(mx, __shfl_xor(mx, 1, 32));
            mx = fmaxf(mx, __shfl_xor(mx, 2, 32));
            mx = fmaxf(mx, __shfl_xor(mx, 4, 32));
            mx = fmaxf(mx, __shfl_xor(mx, 8, 32));
            float mnew = fmaxf(mrun[r], mx);
            float corr = __expf(mrun[r] - mnew);
            mrun[r] = mnew;
            float e0 = __expf(v0 - mnew);
            float e1 = __expf(v1 - mnew);
            float rs = e0 + e1;
            rs += __shfl_xor(rs, 1, 32);
            rs += __shfl_xor(rs, 2, 32);
            rs += __shfl_xor(rs, 4, 32);
            rs += __shfl_xor(rs, 8, 32);
            lrun[r] = lrun[r] * corr + rs;
            p0[r] = e0;
            p1[r] = e1;
#pragma unroll
            for (int j = 0; j < 4; ++j) acco[j][r] *= corr;
        }

        // ---- C-layout -> A-layout for P via per-wave LDS tile ----
#pragma unroll
        for (int r = 0; r < 8; ++r) {
            const int row = r + half * 8;
            plds[wave][row * 32 + l16]      = (bf16)p0[r];
            plds[wave][row * 32 + 16 + l16] = (bf16)p1[r];
        }
        ABf apv;
        load_op(&plds[wave][l16 * 32 + half * 8], apv);

        // ---- O += P @ V (V operands already in flight/registers) ----
#pragma unroll
        for (int j = 0; j < 4; ++j)
            acco[j] = wmma_bf16(apv, bv[j], acco[j]);
    }

    // ---- epilogue: normalize and store to [b, s, h*64 + d] bf16 ----
#pragma unroll
    for (int j = 0; j < 4; ++j) {
#pragma unroll
        for (int r = 0; r < 8; ++r) {
            const int srow = q0 + r + half * 8;
            const int col  = hi * DHEAD + j * 16 + l16;
            const float val = acco[j][r] / lrun[r];
            ob[((size_t)(bi * SEQ + srow)) * D_MODEL + col] = (bf16)val;
        }
    }
}

// ---------------------------- output projection ----------------------------

__global__ __launch_bounds__(256) void out_gemm_kernel(
    const bf16* __restrict__ ab, const bf16* __restrict__ woT,
    const float* __restrict__ bo, float* __restrict__ out)
{
    const int lane = threadIdx.x & 31;
    const int wave = threadIdx.x >> 5;
    const int half = lane >> 4;
    const int l16  = lane & 15;
    const int m0   = blockIdx.x * 16;
    const int n0   = blockIdx.y * 512 + wave * 64;

    const bf16* aptr  = ab  + (size_t)(m0 + l16) * D_MODEL + half * 8;
    const bf16* bptr0 = woT + (size_t)(n0 + 0 * 16 + l16) * D_MODEL + half * 8;
    const bf16* bptr1 = woT + (size_t)(n0 + 1 * 16 + l16) * D_MODEL + half * 8;
    const bf16* bptr2 = woT + (size_t)(n0 + 2 * 16 + l16) * D_MODEL + half * 8;
    const bf16* bptr3 = woT + (size_t)(n0 + 3 * 16 + l16) * D_MODEL + half * 8;

    v8f acc[4];
#pragma unroll
    for (int j = 0; j < 4; ++j) acc[j] = vzero8();

    gemm_16x64_pipe(aptr, bptr0, bptr1, bptr2, bptr3, acc);

#pragma unroll
    for (int j = 0; j < 4; ++j) {
        const int col = n0 + j * 16 + l16;
        const float bcol = bo[col];
#pragma unroll
        for (int r = 0; r < 8; ++r) {
            const int row = m0 + r + half * 8;
            out[(size_t)row * D_MODEL + col] = acc[j][r] + bcol;
        }
    }
}

// ------------------------------- launcher ----------------------------------

extern "C" void kernel_launch(void* const* d_in, const int* in_sizes, int n_in,
                              void* d_out, int out_size, void* d_ws, size_t ws_size,
                              hipStream_t stream) {
    const float* x    = (const float*)d_in[0];
    const float* Wqkv = (const float*)d_in[1];
    const float* bqkv = (const float*)d_in[2];
    const float* Wo   = (const float*)d_in[3];
    const float* bo   = (const float*)d_in[4];
    float* out = (float*)d_out;

    char* p = (char*)d_ws;
    bf16* xb   = (bf16*)p;  p += (size_t)ROWS * D_MODEL * 2;        // 16.8 MB
    bf16* wqT  = (bf16*)p;  p += (size_t)QKV_N * D_MODEL * 2;       //  6.3 MB
    bf16* woT  = (bf16*)p;  p += (size_t)D_MODEL * D_MODEL * 2;     //  2.1 MB
    bf16* qb   = (bf16*)p;  p += (size_t)64 * SEQ * DHEAD * 2;      // 16.8 MB
    bf16* kb   = (bf16*)p;  p += (size_t)64 * SEQ * DHEAD * 2;      // 16.8 MB
    bf16* vtb  = (bf16*)p;  p += (size_t)64 * DHEAD * SEQ * 2;      // 16.8 MB
    bf16* attn = (bf16*)p;  p += (size_t)ROWS * D_MODEL * 2;        // 16.8 MB

    cvt_f32_bf16<<<(ROWS * D_MODEL) / 256, 256, 0, stream>>>(x, xb, ROWS * D_MODEL);
    cvt_transpose<<<(D_MODEL * QKV_N) / 256, 256, 0, stream>>>(Wqkv, wqT, D_MODEL, QKV_N);
    cvt_transpose<<<(D_MODEL * D_MODEL) / 256, 256, 0, stream>>>(Wo, woT, D_MODEL, D_MODEL);

    qkv_gemm_kernel<<<dim3(ROWS / 16, QKV_N / 512), 256, 0, stream>>>(
        xb, wqT, bqkv, qb, kb, vtb);

    attn_kernel<<<dim3(64, SEQ / (16 * 8)), 256, 0, stream>>>(qb, kb, vtb, attn);

    out_gemm_kernel<<<dim3(ROWS / 16, D_MODEL / 512), 256, 0, stream>>>(
        attn, woT, bo, out);
}
